// UMGMQuantizer_49701361550148
// MI455X (gfx1250) — matrix operand order repaired and checked by reading, not compiled, not run on hardware
//
#include <hip/hip_runtime.h>

// MI455X / gfx1250, wave32. Matrix ops via V_WMMA_F32_16X16X32_BF16.
// Forward of the two-level gumbel VQ reduces to: per (n,m) argmax_k of
//   st*(2*<z,c_k> - |c_k|^2) + gumbel(u),  st = max(temp,eps)/sqrt(K)
// (|z|^2 is constant per row -> drops out of argmax; softmax is monotone;
//  straight-through sample == one_hot(argmax) in the forward pass),
// then residual + second level + sum of the two gathered codewords.

typedef __attribute__((ext_vector_type(16))) __bf16 v16bf;
typedef __attribute__((ext_vector_type(8)))  __bf16 v8bf;
typedef __attribute__((ext_vector_type(8)))  float  v8f;

#define N_TOT  4096
#define CH     1024
#define M_CB   8
#define K_CB   2048
#define D_CB   128
#define EPSQ   1e-6f
#define INV_SQRTK 0.022097086912079608f   // 1/sqrt(2048)

#define ROWS   32          // n rows per block (2 WMMA row-tiles sharing B)
#define TILES  2
#define ZSTR   136         // padded bf16 row stride in LDS A-tile (conflict-free b128 reads)
#define WAVES  4
#define TPB    (WAVES * 32)

// ---------------------------------------------------------------------------
// Prepass: fp32 codebook row -> bf16 copy + squared L2 norm. One 128-wide row
// per block (M*K = 16384 blocks per codebook). Cheap: 16 MB total read.
// ---------------------------------------------------------------------------
__global__ void __launch_bounds__(128)
prep_cb(const float* __restrict__ cbf, __bf16* __restrict__ cbb,
        float* __restrict__ c2)
{
    const int row = blockIdx.x;          // 0 .. M*K-1
    const int t   = threadIdx.x;         // 0 .. 127
    const float v = cbf[(size_t)row * D_CB + t];
    cbb[(size_t)row * D_CB + t] = (__bf16)v;

    float s = v * v;
    #pragma unroll
    for (int mk = 16; mk >= 1; mk >>= 1) s += __shfl_xor(s, mk, 32);

    __shared__ float part[4];
    if ((t & 31) == 0) part[t >> 5] = s;
    __syncthreads();
    if (t == 0) c2[row] = part[0] + part[1] + part[2] + part[3];
}

// ---------------------------------------------------------------------------
// Fused two-level quantizer. Block = (32-row n-tile, codebook slot m).
// 4 waves split the K=2048 code dimension; each wave computes TWO 16x16
// logit tiles per k-tile, reusing the same B fragments (halves codebook L2
// traffic, interleaves two independent WMMA chains to cover RAW hazards).
// ---------------------------------------------------------------------------
__global__ void __launch_bounds__(TPB)
umgm_fused(const float* __restrict__ x,
           const float* __restrict__ cb0f, const float* __restrict__ cb1f,
           const __bf16* __restrict__ cb0b, const __bf16* __restrict__ cb1b,
           const float* __restrict__ c20,  const float* __restrict__ c21,
           const float* __restrict__ t0,   const float* __restrict__ t1,
           const float* __restrict__ u0,   const float* __restrict__ u1,
           float* __restrict__ out)
{
    const int m    = blockIdx.y;
    const int n0   = blockIdx.x * ROWS;
    const int tid  = threadIdx.x;
    const int lane = tid & 31;
    const int wave = tid >> 5;
    const int col  = lane & 15;      // WMMA D: N index (code within k-tile)
    const int half = lane >> 4;      // WMMA D: row-half selector

    __shared__ __bf16 At[ROWS * ZSTR] __attribute__((aligned(16)));  // bf16 A tiles
    __shared__ float  xt[ROWS * D_CB];                               // fp32 x, then deq0
    __shared__ int    idxSel[ROWS];
    __shared__ float  redV[WAVES][ROWS];
    __shared__ int    redI[WAVES][ROWS];

    // Stage x slice (read exactly once -> non-temporal): fp32 copy + bf16 A-tiles.
    for (int e = tid; e < ROWS * D_CB; e += TPB) {
        const int r = e >> 7, d = e & 127;
        const float v =
            __builtin_nontemporal_load(&x[(size_t)(n0 + r) * CH + m * D_CB + d]);
        xt[e] = v;
        At[r * ZSTR + d] = (__bf16)v;
    }
    __syncthreads();

    const size_t MK = (size_t)M_CB * K_CB;

    for (int lev = 0; lev < 2; ++lev) {
        const __bf16* cbb = lev ? cb1b : cb0b;
        const float*  c2p = lev ? c21 : c20;
        const float*  up  = lev ? u1  : u0;
        const float   tv  = lev ? t1[m] : t0[m];
        const float   st  = fmaxf(tv, EPSQ) * INV_SQRTK;

        // A fragments for both row-tiles. ISA 16-bit A layout per 32-wide chunk:
        // lanes 0-15 hold K {0..7,16..23}+32c, lanes 16-31 hold K {8..15,24..31}+32c.
        v16bf a[TILES][4];
        {
            const int s0 = half * 8;
            #pragma unroll
            for (int t2 = 0; t2 < TILES; ++t2) {
                const int arow = t2 * 16 + col;
                #pragma unroll
                for (int c = 0; c < 4; ++c) {
                    v8bf lo = *(const v8bf*)&At[arow * ZSTR + 32 * c + s0];
                    v8bf hs = *(const v8bf*)&At[arow * ZSTR + 32 * c + 16 + s0];
                    a[t2][c] = __builtin_shufflevector(lo, hs, 0,1,2,3,4,5,6,7,
                                                               8,9,10,11,12,13,14,15);
                }
            }
        }

        float bestV[TILES][8];
        int   bestI[TILES][8];
        #pragma unroll
        for (int t2 = 0; t2 < TILES; ++t2)
            #pragma unroll
            for (int r = 0; r < 8; ++r) { bestV[t2][r] = -3.4e38f; bestI[t2][r] = 0; }

        const float* urow0 = up + (size_t)(n0 + 8 * half) * MK + (size_t)m * K_CB;

        // Uniform per-wave loop: EXEC stays all-ones around every WMMA.
        for (int kt = wave; kt < K_CB / 16; kt += WAVES) {
            const int code = kt * 16 + col;
            // B fragments: column = code, contraction d matches A layout:
            // lanes 0-15 d = 32c+{0..7,16..23}, lanes 16-31 d = 32c+{8..15,24..31}.
            const __bf16* bp = cbb + ((size_t)m * K_CB + code) * D_CB + half * 8;
            v16bf bfr[4];
            #pragma unroll
            for (int c = 0; c < 4; ++c) {
                v8bf blo = *(const v8bf*)(bp + 32 * c);
                v8bf bhi = *(const v8bf*)(bp + 32 * c + 16);
                bfr[c] = __builtin_shufflevector(blo, bhi, 0,1,2,3,4,5,6,7,
                                                           8,9,10,11,12,13,14,15);
            }

            // Two independent accumulator chains share the B fragments.
            v8f acc[TILES];
            #pragma unroll
            for (int t2 = 0; t2 < TILES; ++t2) acc[t2] = (v8f){};
            #pragma unroll
            for (int c = 0; c < 4; ++c)
                #pragma unroll
                for (int t2 = 0; t2 < TILES; ++t2)
                    acc[t2] = __builtin_amdgcn_wmma_f32_16x16x32_bf16(
                                  false, a[t2][c], false, bfr[c],
                                  (short)0, acc[t2], false, false);

            const float negc2 = -st * c2p[(size_t)m * K_CB + code];
            #pragma unroll
            for (int t2 = 0; t2 < TILES; ++t2) {
                const float* uc = urow0 + (size_t)(t2 * 16) * MK + code;
                #pragma unroll
                for (int r = 0; r < 8; ++r) {
                    float uv = __builtin_nontemporal_load(&uc[(size_t)r * MK]);
                    uv = fminf(fmaxf(uv, EPSQ), 1.0f - EPSQ);
                    const float g   = -__logf(-__logf(uv));
                    const float val = fmaf(2.0f * st, acc[t2][r], negc2) + g;
                    if (val > bestV[t2][r]) { bestV[t2][r] = val; bestI[t2][r] = code; }
                }
            }
        }

        // Argmax reduce inside each 16-lane group (row r+8*half, code varies by lane).
        #pragma unroll
        for (int t2 = 0; t2 < TILES; ++t2) {
            #pragma unroll
            for (int r = 0; r < 8; ++r) {
                #pragma unroll
                for (int mk = 8; mk >= 1; mk >>= 1) {
                    const float ov = __shfl_xor(bestV[t2][r], mk, 32);
                    const int   oi = __shfl_xor(bestI[t2][r], mk, 32);
                    if (ov > bestV[t2][r] ||
                        (ov == bestV[t2][r] && oi < bestI[t2][r])) {
                        bestV[t2][r] = ov; bestI[t2][r] = oi;
                    }
                }
                if (col == 0) {
                    redV[wave][t2 * 16 + r + 8 * half] = bestV[t2][r];
                    redI[wave][t2 * 16 + r + 8 * half] = bestI[t2][r];
                }
            }
        }
        __syncthreads();
        if (tid < ROWS) {
            float bv = redV[0][tid]; int bi = redI[0][tid];
            #pragma unroll
            for (int w = 1; w < WAVES; ++w) {
                const float ov = redV[w][tid]; const int oi = redI[w][tid];
                if (ov > bv || (ov == bv && oi < bi)) { bv = ov; bi = oi; }
            }
            idxSel[tid] = bi;
        }
        __syncthreads();

        if (lev == 0) {
            // Gather deq0, form residual A-tiles for level 1; xt now holds deq0.
            for (int e = tid; e < ROWS * D_CB; e += TPB) {
                const int r = e >> 7, d = e & 127;
                const float dq = cb0f[((size_t)m * K_CB + idxSel[r]) * D_CB + d];
                At[r * ZSTR + d] = (__bf16)(xt[e] - dq);
                xt[e] = dq;
            }
            __syncthreads();
        } else {
            // xHat = deq0 + deq1 (written exactly once -> non-temporal store).
            for (int e = tid; e < ROWS * D_CB; e += TPB) {
                const int r = e >> 7, d = e & 127;
                const float dq1 = cb1f[((size_t)m * K_CB + idxSel[r]) * D_CB + d];
                __builtin_nontemporal_store(
                    xt[e] + dq1, &out[(size_t)(n0 + r) * CH + m * D_CB + d]);
            }
        }
    }
}

// ---------------------------------------------------------------------------
extern "C" void kernel_launch(void* const* d_in, const int* in_sizes, int n_in,
                              void* d_out, int out_size, void* d_ws, size_t ws_size,
                              hipStream_t stream)
{
    (void)in_sizes; (void)n_in; (void)out_size; (void)ws_size;

    const float* x   = (const float*)d_in[0];
    const float* cb0 = (const float*)d_in[1];
    const float* cb1 = (const float*)d_in[2];
    const float* t0  = (const float*)d_in[3];
    const float* t1  = (const float*)d_in[4];
    const float* u0  = (const float*)d_in[5];
    const float* u1  = (const float*)d_in[6];
    float* out = (float*)d_out;

    // Workspace layout: cb0 bf16 (4MB) | cb1 bf16 (4MB) | c2_0 (64KB) | c2_1 (64KB)
    char* ws = (char*)d_ws;
    const size_t CB_ELTS = (size_t)M_CB * K_CB * D_CB;
    __bf16* cb0b = (__bf16*)ws;
    __bf16* cb1b = (__bf16*)(ws + CB_ELTS * sizeof(__bf16));
    float*  c20  = (float*)(ws + 2 * CB_ELTS * sizeof(__bf16));
    float*  c21  = c20 + (size_t)M_CB * K_CB;

    prep_cb<<<M_CB * K_CB, 128, 0, stream>>>(cb0, cb0b, c20);
    prep_cb<<<M_CB * K_CB, 128, 0, stream>>>(cb1, cb1b, c21);

    dim3 grid(N_TOT / ROWS, M_CB);
    umgm_fused<<<grid, TPB, 0, stream>>>(x, cb0, cb1, cb0b, cb1b,
                                         c20, c21, t0, t1, u0, u1, out);
}
